// StochasticDurationPredictor_64338610094488
// MI455X (gfx1250) — compile-verified, matrix operand order checked by reading, not run on hardware
//
#include <hip/hip_runtime.h>
#include <math.h>

// ---------------------------------------------------------------------------
// Shapes
// ---------------------------------------------------------------------------
static constexpr int kB  = 64;
static constexpr int kT  = 1024;
static constexpr int kF  = 192;
static constexpr int kM  = kB * kT;   // 65536 token rows

typedef __attribute__((ext_vector_type(16))) _Float16 v16h;
typedef __attribute__((ext_vector_type(8)))  float    v8f;
typedef int v4i __attribute__((vector_size(16)));

#if defined(__has_builtin)
#if __has_builtin(__builtin_amdgcn_global_load_async_to_lds_b128)
#define USE_ASYNC_LDS 1
#endif
#endif

// ---------------------------------------------------------------------------
// Device helpers
// ---------------------------------------------------------------------------
__device__ __forceinline__ float gelu_tanh(float x) {
  float x3 = x * x * x;
  return 0.5f * x * (1.0f + tanhf(0.7978845608028654f * (x + 0.044715f * x3)));
}
__device__ __forceinline__ float softplusf(float x) {
  return (x > 20.0f) ? x : log1pf(expf(x));
}
__device__ __forceinline__ void wait_async0() {
#if defined(__has_builtin)
#if __has_builtin(__builtin_amdgcn_s_wait_asynccnt)
  __builtin_amdgcn_s_wait_asynccnt(0);
#else
  asm volatile("s_wait_asynccnt 0x0" ::: "memory");
#endif
#else
  asm volatile("s_wait_asynccnt 0x0" ::: "memory");
#endif
}

// ---------------------------------------------------------------------------
// WMMA GEMM: C[M x N] = A[M x K] @ W[K x N] + bias (+ addsrc).
// A path 1: f16 mirror A16 -> LDS via async b128 copies (identity layout).
// A path 2: f32 A -> convert to f16 in LDS.
// W always f32 -> converted/transposed into LDS.
// Block: 256 threads (8 waves). Tile: 64 (M) x 64 (N). K <= 192 (here 192).
// ---------------------------------------------------------------------------
__global__ __launch_bounds__(256)
void gemm_wmma_kernel(const float* __restrict__ A,
                      const _Float16* __restrict__ A16, int lda,
                      const float* __restrict__ W, int ldw,
                      const float* __restrict__ bias,
                      const float* __restrict__ addsrc,
                      float* __restrict__ C, int ldc,
                      int M, int N, int K)
{
  __shared__ _Float16 As[64][192];   // [m][k], unpadded: identity layout for async copy
  __shared__ _Float16 Bs[64][200];   // [n][k], padded (W transposed)

  const int tid = threadIdx.x;
  const int m0  = blockIdx.y * 64;
  const int n0  = blockIdx.x * 64;

  bool used_async = false;
  if (A16) {
    // prefetch next M-slab (emits global_prefetch_b8)
    if (tid == 0 && (m0 + 64) < M) {
      __builtin_prefetch(&A16[(size_t)(m0 + 64) * lda], 0, 1);
    }
#if defined(USE_ASYNC_LDS)
    // 64 rows x 192 halves = 24576 bytes, contiguous in both global and LDS.
    {
      auto gsrc = (__attribute__((address_space(1))) v4i*)(A16 + (size_t)m0 * lda);
      auto ldst = (__attribute__((address_space(3))) v4i*)(&As[0][0]);
      for (int idx = tid; idx < 1536; idx += 256) {   // 1536 x 16B
        __builtin_amdgcn_global_load_async_to_lds_b128(gsrc + idx, ldst + idx, 0, 0);
      }
      used_async = true;
    }
#else
    {
      _Float16* asf = &As[0][0];
      const _Float16* src = A16 + (size_t)m0 * lda;
      for (int idx = tid; idx < 64 * 192; idx += 256) asf[idx] = src[idx];
    }
#endif
  } else {
    if (tid == 0 && (m0 + 64) < M) {
      __builtin_prefetch(&A[(size_t)(m0 + 64) * lda], 0, 1);
    }
    for (int idx = tid; idx < 64 * K; idx += 256) {
      int r = idx / K, c = idx % K;
      int m = m0 + r;
      As[r][c] = (_Float16)((m < M) ? A[(size_t)m * lda + c] : 0.0f);
    }
  }

  // Stage W (K x 64) transposed -> Bs[n][k], zero-pad columns beyond N
  for (int idx = tid; idx < 64 * K; idx += 256) {
    int k = idx >> 6, n = idx & 63;
    float v = ((n0 + n) < N) ? W[(size_t)k * ldw + (n0 + n)] : 0.0f;
    Bs[n][k] = (_Float16)v;
  }

  if (used_async) wait_async0();
  __syncthreads();

  const int wave  = tid >> 5;
  const int lane  = tid & 31;
  const int mt    = (wave & 3) * 16;        // 4 M-subtiles
  const int ntb   = (wave >> 2) * 32;       // 2 N-groups of 32 (two 16-tiles)
  const int l15   = lane & 15;
  const int lhalf = (lane & 16) >> 1;       // 0 or 8 (K-half select per lane)

  v8f c0 = {}; v8f c1 = {};
  for (int kb = 0; kb < K; kb += 32) {
    v16h a, b0v, b1v;
#pragma unroll
    for (int h = 0; h < 16; ++h) {
      // 16-bit A/B fragment layout: halves 0..7 -> K 0..7 (lane<16) / 8..15,
      // halves 8..15 -> K 16..23 / 24..31
      int kk = kb + ((h & 8) << 1) + lhalf + (h & 7);
      a[h]   = As[mt + l15][kk];
      b0v[h] = Bs[ntb + l15][kk];
      b1v[h] = Bs[ntb + 16 + l15][kk];
    }
    c0 = __builtin_amdgcn_wmma_f32_16x16x32_f16(false, a, false, b0v,
                                                (short)0, c0, false, false);
    c1 = __builtin_amdgcn_wmma_f32_16x16x32_f16(false, a, false, b1v,
                                                (short)0, c1, false, false);
  }

  // C/D layout: lane 0-15 -> N=lane, M=vgpr r; lane 16-31 -> N=lane-16, M=8+r
  const int mrow = m0 + mt + ((lane >> 4) << 3);
#pragma unroll
  for (int r = 0; r < 8; ++r) {
    int m = mrow + r;
    if (m < M) {
      int n = n0 + ntb + l15;
      if (n < N) {
        float v = c0[r] + (bias ? bias[n] : 0.0f);
        if (addsrc) v += addsrc[(size_t)m * ldc + n];
        C[(size_t)m * ldc + n] = v;
      }
      n += 16;
      if (n < N) {
        float v = c1[r] + (bias ? bias[n] : 0.0f);
        if (addsrc) v += addsrc[(size_t)m * ldc + n];
        C[(size_t)m * ldc + n] = v;
      }
    }
  }
}

// ---------------------------------------------------------------------------
// Depthwise conv (K=3, dilated) * mask + bias, then LayerNorm + gelu.
// One block per token row, 192 threads (one per channel).
// Writes f32 output plus an f16 mirror (GEMM A operand).
// ---------------------------------------------------------------------------
__global__ __launch_bounds__(192)
void dw_ln_gelu_kernel(const float* __restrict__ X, const float* __restrict__ mask,
                       const float* __restrict__ sw, const float* __restrict__ sb,
                       const float* __restrict__ s1, const float* __restrict__ b1,
                       float* __restrict__ Y, _Float16* __restrict__ Y16, int dil)
{
  const int m = blockIdx.x;
  const int b = m >> 10;
  const int t = m & 1023;
  const int c = threadIdx.x;

  float acc = sb[c];
#pragma unroll
  for (int k = 0; k < 3; ++k) {
    int tt = t + (k - 1) * dil;
    if (tt >= 0 && tt < kT) {
      int mm = (b << 10) + tt;
      acc += X[(size_t)mm * kF + c] * mask[mm] * sw[k * kF + c];
    }
  }

  __shared__ float rs[192], rq[192];
  rs[c] = acc; rq[c] = acc * acc;
  __syncthreads();
  for (int s = 96; s >= 3; s >>= 1) {
    if (c < s) { rs[c] += rs[c + s]; rq[c] += rq[c + s]; }
    __syncthreads();
  }
  float sum = rs[0] + rs[1] + rs[2];
  float sq  = rq[0] + rq[1] + rq[2];
  float mean = sum * (1.0f / 192.0f);
  float var  = sq * (1.0f / 192.0f) - mean * mean;
  float yn = (acc - mean) * rsqrtf(var + 1e-6f) * s1[c] + b1[c];
  float yo = gelu_tanh(yn);
  Y[(size_t)m * kF + c] = yo;
  Y16[(size_t)m * kF + c] = (_Float16)yo;
}

// LN + gelu + residual add: Xres += gelu(LN(T2)); also refresh f16 mirror.
__global__ __launch_bounds__(192)
void ln_gelu_res_kernel(const float* __restrict__ T2,
                        const float* __restrict__ s2, const float* __restrict__ b2,
                        float* __restrict__ Xres, _Float16* __restrict__ X16)
{
  const int m = blockIdx.x;
  const int c = threadIdx.x;
  float v = T2[(size_t)m * kF + c];

  __shared__ float rs[192], rq[192];
  rs[c] = v; rq[c] = v * v;
  __syncthreads();
  for (int s = 96; s >= 3; s >>= 1) {
    if (c < s) { rs[c] += rs[c + s]; rq[c] += rq[c + s]; }
    __syncthreads();
  }
  float sum = rs[0] + rs[1] + rs[2];
  float sq  = rq[0] + rq[1] + rq[2];
  float mean = sum * (1.0f / 192.0f);
  float var  = sq * (1.0f / 192.0f) - mean * mean;
  float yn = (v - mean) * rsqrtf(var + 1e-6f) * s2[c] + b2[c];
  float nv = Xres[(size_t)m * kF + c] + gelu_tanh(yn);
  Xres[(size_t)m * kF + c] = nv;
  X16[(size_t)m * kF + c]  = (_Float16)nv;
}

// Outer product (1 -> F): Y[m,c] = z0[m]*wrow[c] + bias[c] (+ g[m,c])
__global__ __launch_bounds__(256)
void outer1_kernel(const float* __restrict__ z0, const float* __restrict__ wrow,
                   const float* __restrict__ bias, const float* __restrict__ g,
                   float* __restrict__ Y, int M)
{
  int i = blockIdx.x * 256 + threadIdx.x;
  if (i >= M * kF) return;
  int m = i / kF, c = i % kF;
  float v = z0[m] * wrow[c] + bias[c];
  if (g) v += g[i];
  Y[i] = v;
}

// Affine flow entry: z = (aff_m + exp(aff_logs)*z) * mask  (2 channels)
__global__ __launch_bounds__(256)
void affine_kernel(const float* __restrict__ in0, int s0,
                   const float* __restrict__ in1, int s1,
                   const float* __restrict__ am, const float* __restrict__ alogs,
                   const float* __restrict__ mask,
                   float* __restrict__ z0, float* __restrict__ z1)
{
  int m = blockIdx.x * 256 + threadIdx.x;
  if (m >= kM) return;
  float msk = mask[m];
  float a0 = am[0], a1 = am[1];
  float e0 = expf(alogs[0]), e1 = expf(alogs[1]);
  float v0 = in0[(size_t)m * s0];
  float v1 = in1[(size_t)m * s1];
  z0[m] = (a0 + e0 * v0) * msk;
  z1[m] = (a1 + e1 * v1) * msk;
}

// ---------------------------------------------------------------------------
// Rational-quadratic spline (NB=10 bins, TB=5): z1 <- spline(z1; h), and
// accumulate log|det| per batch.
// ---------------------------------------------------------------------------
__global__ __launch_bounds__(256)
void spline_kernel(const float* __restrict__ H, int ldh,
                   float* __restrict__ Z1, const float* __restrict__ mask,
                   float* __restrict__ accLd)
{
  const int m = blockIdx.x * 256 + threadIdx.x;
  const float msk = mask[m];
  const float x = Z1[m];
  const float* h = H + (size_t)m * ldh;
  const float invs = 0.07216878364870323f;  // 1/sqrt(192)
  const float TB = 5.0f;

  float uw[10], uh[10];
  float mxw = -1e30f, mxh = -1e30f;
#pragma unroll
  for (int j = 0; j < 10; ++j) {
    uw[j] = h[j] * invs;       mxw = fmaxf(mxw, uw[j]);
    uh[j] = h[10 + j] * invs;  mxh = fmaxf(mxh, uh[j]);
  }
  float sw = 0.f, sh = 0.f;
#pragma unroll
  for (int j = 0; j < 10; ++j) {
    uw[j] = expf(uw[j] - mxw); sw += uw[j];
    uh[j] = expf(uh[j] - mxh); sh += uh[j];
  }
  float cw[11], ch[11];
  cw[0] = 0.f; ch[0] = 0.f;
#pragma unroll
  for (int j = 0; j < 10; ++j) {
    float wj = 0.001f + 0.99f * uw[j] / sw;
    float hj = 0.001f + 0.99f * uh[j] / sh;
    cw[j + 1] = cw[j] + wj;
    ch[j + 1] = ch[j] + hj;
  }
#pragma unroll
  for (int j = 0; j < 11; ++j) {
    cw[j] = 2.f * TB * cw[j] - TB;
    ch[j] = 2.f * TB * ch[j] - TB;
  }
  cw[0] = -TB; cw[10] = TB; ch[0] = -TB; ch[10] = TB;

  const float cst = 0.5413248546129181f;  // log(exp(1)-1)
  float dv[11];
  dv[0] = 0.001f + softplusf(cst);
  dv[10] = dv[0];
#pragma unroll
  for (int j = 0; j < 9; ++j) dv[j + 1] = 0.001f + softplusf(h[20 + j]);

  bool inside = (x >= -TB) && (x <= TB);
  float xi = fminf(fmaxf(x, -TB), TB);
  int cnt = 0;
#pragma unroll
  for (int j = 0; j < 11; ++j) cnt += (xi >= cw[j]) ? 1 : 0;
  int idx = cnt - 1;
  idx = idx < 0 ? 0 : (idx > 9 ? 9 : idx);

  float icw = cw[idx], ibw = cw[idx + 1] - cw[idx];
  float ich = ch[idx], ih  = ch[idx + 1] - ch[idx];
  float dlt = ih / ibw;
  float d0 = dv[idx], d1 = dv[idx + 1];
  float th = (xi - icw) / ibw;
  float tt = th * (1.f - th);
  float num = ih * (dlt * th * th + d0 * tt);
  float den = dlt + (d0 + d1 - 2.f * dlt) * tt;
  float outv = ich + num / den;
  float dnum = dlt * dlt * (d1 * th * th + 2.f * dlt * tt + d0 * (1.f - th) * (1.f - th));
  float lad = logf(dnum) - 2.f * logf(den);

  Z1[m] = (inside ? outv : x) * msk;
  float ladv = (inside ? lad : 0.f) * msk;

  __shared__ float red[256];
  red[threadIdx.x] = ladv;
  __syncthreads();
  for (int s = 128; s > 0; s >>= 1) {
    if (threadIdx.x < s) red[threadIdx.x] += red[threadIdx.x + s];
    __syncthreads();
  }
  if (threadIdx.x == 0) atomicAdd(&accLd[(blockIdx.x * 256) >> 10], red[0]);
}

// Between the two flow stacks: u=sigmoid(zq0); z0=log(clip(w-u)); logq/logdet.
__global__ __launch_bounds__(256)
void mid_kernel(const float* __restrict__ zq0, const float* __restrict__ zq1,
                const float* __restrict__ wdur, const float* __restrict__ mask,
                float* __restrict__ z0out, float* __restrict__ z1out,
                float* __restrict__ accLogq, float* __restrict__ accLd)
{
  const float LOG2PI = 1.8378770664093453f;
  int m = blockIdx.x * 256 + threadIdx.x;
  float msk = mask[m];
  float u   = (1.0f / (1.0f + expf(-zq0[m]))) * msk;
  float z0v = (wdur[m] - u) * msk;
  float z1v = zq1[m];
  float logq_c = -0.5f * (LOG2PI + z1v * z1v) * msk;
  float z0l = logf(fmaxf(z0v, 1e-5f)) * msk;
  z0out[m] = z0l;
  z1out[m] = z1v;

  __shared__ float r1[256], r2[256];
  r1[threadIdx.x] = logq_c;
  r2[threadIdx.x] = -z0l;
  __syncthreads();
  for (int s = 128; s > 0; s >>= 1) {
    if (threadIdx.x < s) {
      r1[threadIdx.x] += r1[threadIdx.x + s];
      r2[threadIdx.x] += r2[threadIdx.x + s];
    }
    __syncthreads();
  }
  if (threadIdx.x == 0) {
    int b = (blockIdx.x * 256) >> 10;
    atomicAdd(&accLogq[b], r1[0]);
    atomicAdd(&accLd[b],   r2[0]);
  }
}

__global__ __launch_bounds__(256)
void final_kernel(const float* __restrict__ z0, const float* __restrict__ z1,
                  const float* __restrict__ mask, float* __restrict__ accN)
{
  const float LOG2PI = 1.8378770664093453f;
  int m = blockIdx.x * 256 + threadIdx.x;
  float msk = mask[m];
  float a = z0[m], b = z1[m];
  float c = -0.5f * (2.f * LOG2PI + a * a + b * b) * msk;

  __shared__ float red[256];
  red[threadIdx.x] = c;
  __syncthreads();
  for (int s = 128; s > 0; s >>= 1) {
    if (threadIdx.x < s) red[threadIdx.x] += red[threadIdx.x + s];
    __syncthreads();
  }
  if (threadIdx.x == 0) atomicAdd(&accN[(blockIdx.x * 256) >> 10], red[0]);
}

__global__ void zero_kernel(float* __restrict__ acc) {
  acc[threadIdx.x] = 0.0f;  // 256 floats: accLd, accLq, accN, accD
}

__global__ __launch_bounds__(256)
void msum_kernel(const float* __restrict__ mask, float* __restrict__ msum) {
  int b = blockIdx.x;
  float s = 0.f;
  for (int t = threadIdx.x; t < kT; t += 256) s += mask[b * kT + t];
  __shared__ float red[256];
  red[threadIdx.x] = s;
  __syncthreads();
  for (int st = 128; st > 0; st >>= 1) {
    if (threadIdx.x < st) red[threadIdx.x] += red[threadIdx.x + st];
    __syncthreads();
  }
  if (threadIdx.x == 0) msum[b] = red[0];
}

__global__ void addaff_kernel(const float* __restrict__ msum,
                              const float* __restrict__ alogs,
                              float* __restrict__ acc) {
  int b = threadIdx.x;
  if (b < kB) acc[b] += msum[b] * (alogs[0] + alogs[1]);
}

__global__ void out_kernel(const float* __restrict__ accN,
                           const float* __restrict__ accLd,
                           const float* __restrict__ accLq,
                           float* __restrict__ out) {
  int b = threadIdx.x;
  if (b < kB) out[b] = accN[b] - accLd[b] + accLq[b];
}

// ---------------------------------------------------------------------------
// Host orchestration
// ---------------------------------------------------------------------------
extern "C" void kernel_launch(void* const* d_in, const int* in_sizes, int n_in,
                              void* d_out, int out_size, void* d_ws, size_t ws_size,
                              hipStream_t stream)
{
  (void)in_sizes; (void)n_in; (void)out_size; (void)ws_size;
  auto P = [&](int i) { return (const float*)d_in[i]; };

  const int M = kM;
  float* ws = (float*)d_ws;
  const size_t MF = (size_t)M * kF;
  float* bufH  = ws;              // residual stream
  float* bufT1 = ws + 1 * MF;     // depthwise/LN output
  float* bufT2 = ws + 2 * MF;     // pointwise GEMM output
  float* bufG1 = ws + 3 * MF;     // x projection (g for 'flows')
  float* bufG2 = ws + 4 * MF;     // x + h_w (g for 'post_flows')
  float* bufP  = ws + 5 * MF;     // M x 32 spline-param projection
  float* zq0   = bufP + (size_t)M * 32;
  float* zq1   = zq0 + M;
  float* zf0   = zq0 + 2 * (size_t)M;
  float* zf1   = zq0 + 3 * (size_t)M;
  float* accLd = zq0 + 4 * (size_t)M;
  float* accLq = accLd + 64;
  float* accN  = accLd + 128;
  float* accD  = accLd + 192;
  float* msum  = accLd + 256;
  _Float16* bufT1h = (_Float16*)(accLd + 320);  // f16 mirror of bufT1
  _Float16* bufHh  = bufT1h + MF;               // f16 mirror of bufH

  const float* X    = P(0);
  const float* MASK = P(1);
  const float* WD   = P(2);
  const float* EQ   = P(3);
  // params pytree (sorted dict keys) starts at index 4:
  // convs:4..11  flows:12..61  post_convs:62..69  post_flows:70..119
  // post_pre_b:120 post_pre_w:121 post_proj_b:122 post_proj_w:123
  // pre_b:124 pre_w:125 proj_b:126 proj_w:127

  struct DDSp { const float *ln1_b,*ln1_s,*ln2_b,*ln2_s,*p1_b,*p1_w,*sep_b,*sep_w; };
  auto ddsAt = [&](int base) {
    DDSp d{P(base), P(base+1), P(base+2), P(base+3),
           P(base+4), P(base+5), P(base+6), P(base+7)};
    return d;
  };

  auto gemm = [&](const float* A, const _Float16* A16, int lda,
                  const float* W, int ldw,
                  const float* bias, const float* add, float* C, int ldc, int N) {
    dim3 g((N + 63) / 64, M / 64);
    gemm_wmma_kernel<<<g, 256, 0, stream>>>(A, A16, lda, W, ldw, bias, add,
                                            C, ldc, M, N, kF);
  };

  auto runDDS = [&](DDSp d, float* Xres, _Float16* Xres16) {
    int dil = 1;
    for (int i = 0; i < 3; ++i) {
      dw_ln_gelu_kernel<<<M, 192, 0, stream>>>(Xres, MASK,
          d.sep_w + (size_t)i * 3 * kF, d.sep_b + (size_t)i * kF,
          d.ln1_s + (size_t)i * kF, d.ln1_b + (size_t)i * kF,
          bufT1, bufT1h, dil);
      gemm(bufT1, bufT1h, kF, d.p1_w + (size_t)i * kF * kF, kF,
           d.p1_b + (size_t)i * kF, nullptr, bufT2, kF, kF);
      ln_gelu_res_kernel<<<M, 192, 0, stream>>>(bufT2,
          d.ln2_s + (size_t)i * kF, d.ln2_b + (size_t)i * kF, Xres, Xres16);
      dil *= 3;
    }
  };

  auto runFlowStack = [&](int base, const float* g,
                          const float* in0, int s0, const float* in1, int s1,
                          float* za, float* zb, float* acc,
                          float** outz0, float** outz1) {
    const float* alogs = P(base);
    const float* am    = P(base + 1);
    affine_kernel<<<M / 256, 256, 0, stream>>>(in0, s0, in1, s1, am, alogs, MASK, za, zb);
    float* pz0 = za; float* pz1 = zb;
    for (int f = 0; f < 4; ++f) {
      int cb = base + 2 + f * 12;   // convflow leaves: dds(8), pre_b, pre_w, proj_b, proj_w
      DDSp d = ddsAt(cb);
      const float* pre_b  = P(cb + 8);
      const float* pre_w  = P(cb + 9);
      const float* proj_b = P(cb + 10);
      const float* proj_w = P(cb + 11);
      outer1_kernel<<<(M * kF + 255) / 256, 256, 0, stream>>>(pz0, pre_w, pre_b, g, bufH, M);
      runDDS(d, bufH, bufHh);
      gemm(bufH, bufHh, kF, proj_w, 29, proj_b, nullptr, bufP, 32, 29);
      spline_kernel<<<M / 256, 256, 0, stream>>>(bufP, 32, pz1, MASK, acc);
      float* tmp = pz0; pz0 = pz1; pz1 = tmp;   // z[..., ::-1]
    }
    *outz0 = pz0; *outz1 = pz1;
  };

  // --- schedule ---
  zero_kernel<<<1, 256, 0, stream>>>(accLd);
  msum_kernel<<<kB, 256, 0, stream>>>(MASK, msum);

  // main x path: pre -> dds -> proj  => bufG1
  gemm(X, nullptr, kF, P(125), kF, P(124), nullptr, bufH, kF, kF);
  runDDS(ddsAt(4), bufH, bufHh);
  gemm(bufH, bufHh, kF, P(127), kF, P(126), nullptr, bufG1, kF, kF);

  // w path: post_pre -> post_dds -> post_proj (+x) => bufG2
  outer1_kernel<<<(M * kF + 255) / 256, 256, 0, stream>>>(WD, P(121), P(120), nullptr, bufH, M);
  runDDS(ddsAt(62), bufH, bufHh);
  gemm(bufH, bufHh, kF, P(123), kF, P(122), bufG1, bufG2, kF, kF);

  // post flow stack on e_q (interleaved stride-2), g = x + h_w
  float *pz0, *pz1;
  runFlowStack(70, bufG2, EQ, 2, EQ + 1, 2, zq0, zq1, accD, &pz0, &pz1);

  // mid: u = sigmoid(z_u); z0 = log(clip(w-u)); logq / logdet contributions
  mid_kernel<<<M / 256, 256, 0, stream>>>(pz0, pz1, WD, MASK, zf0, zf1, accLq, accLd);
  addaff_kernel<<<1, 64, 0, stream>>>(msum, P(12), accLd);

  // main flow stack, g = x
  runFlowStack(12, bufG1, zf0, 1, zf1, 1, zf0, zf1, accLd, &pz0, &pz1);

  final_kernel<<<M / 256, 256, 0, stream>>>(pz0, pz1, MASK, accN);
  out_kernel<<<1, 64, 0, stream>>>(accN, accLd, accLq, (float*)d_out);
}
